// WindowCrossAttnFusion_69810398429826
// MI455X (gfx1250) — matrix operand order repaired and checked by reading, not compiled, not run on hardware
//
#include <hip/hip_runtime.h>
#include <hip/hip_bf16.h>

typedef __attribute__((ext_vector_type(16))) _Float16 v16h;
typedef __attribute__((ext_vector_type(8)))  float    v8f;

#define LDX 68   // f32 LDS row stride (floats)
#define LDH 72   // f16 LDS row stride (halves)

__device__ __forceinline__ v8f wmma_f16(v16h a, v16h b, v8f c) {
  // D = A(16x32 f16) * B(32x16 f16) + C(16x16 f32)
  return __builtin_amdgcn_wmma_f32_16x16x32_f16(false, a, false, b, (short)0, c, false, false);
}

// A fragment (16x32, M x K) from f16 LDS tile [row][ld]
__device__ __forceinline__ v16h frag_a_h(const _Float16* S, int ld, int mBase, int kBase, int lane) {
  v16h a;
  int m  = mBase + (lane & 15);
  int k0 = kBase + ((lane >= 16) ? 16 : 0);
#pragma unroll
  for (int j = 0; j < 16; ++j) {
    int k = k0 + j;
    a[j] = S[m * ld + k];
  }
  return a;
}

// A fragment from f32 LDS tile (convert on the fly)
__device__ __forceinline__ v16h frag_a_f(const float* S, int ld, int mBase, int kBase, int lane) {
  v16h a;
  int m  = mBase + (lane & 15);
  int k0 = kBase + ((lane >= 16) ? 16 : 0);
#pragma unroll
  for (int j = 0; j < 16; ++j) {
    int k = k0 + j;
    a[j] = (_Float16)S[m * ld + k];
  }
  return a;
}

// B fragment (32x16, K x N) from global f32 weight, row-major [N][ldw] (N = out-ch rows)
__device__ __forceinline__ v16h frag_b_w(const float* Wm, int ldw, int nBase, int kBase, int lane) {
  v16h b;
  const float* row = Wm + (nBase + (lane & 15)) * ldw + kBase + ((lane >= 16) ? 16 : 0);
#pragma unroll
  for (int j = 0; j < 16; ++j) b[j] = (_Float16)row[j];
  return b;
}

__global__ __launch_bounds__(256)
void wca_fused_gfx1250(const float* __restrict__ f_dem, const float* __restrict__ f_ae,
                       const float* __restrict__ trust,
                       const float* __restrict__ Wq, const float* __restrict__ Wk,
                       const float* __restrict__ Wv, const float* __restrict__ Wo,
                       const float* __restrict__ Wg, const float* __restrict__ bg,
                       const int* __restrict__ alphaP, float* __restrict__ out)
{
  constexpr int H = 256, W = 256, WS = 8;

  __shared__ float    Xdem[64][LDX];     // f_dem window tile [tok][ch] (f32: GEMM A + residual)
  __shared__ float    Xae [64][LDX];     // f_ae window tile; reused later as out staging [oc][tok]
  __shared__ float    Xtr [64][8];       // trust [tok][ch], ch 4..7 zero pad
  __shared__ _Float16 Qs  [64][LDH];     // q [tok][dm]
  __shared__ _Float16 Ksm [64][LDH];     // k [tok][dm]
  __shared__ _Float16 Vsm [64][LDH];     // v [tok][dm]
  __shared__ _Float16 Attn[4][64][LDH];  // per-head softmaxed attention [q][k]
  __shared__ _Float16 Osm [64][LDH];     // attention output [tok][dm]
  __shared__ float    Aa  [64][LDX];     // a = o @ Wo^T [tok][oc] (f32)

  const int t    = threadIdx.x;
  const int lane = t & 31;
  const int wave = t >> 5;
  const int wid  = blockIdx.x;         // 8192 windows: b(8) x wy(32) x wx(32)
  const int b    = wid >> 10;
  const int wy   = (wid >> 5) & 31;
  const int wx   = wid & 31;
  const int hy   = wy * WS, hx = wx * WS;

  // ---------------- phase 0: load window tiles (coalesced 32B rows) -------------
#pragma unroll
  for (int i = 0; i < 2; ++i) {
    int job = t + 256 * i;               // 512 jobs = 64ch x 8 rows
    int c = job >> 3, y = job & 7;
    const float4* sd = (const float4*)(f_dem + (((b * 64 + c) * H + hy + y) * W + hx));
    const float4* sa = (const float4*)(f_ae  + (((b * 64 + c) * H + hy + y) * W + hx));
    float4 d0 = sd[0], d1 = sd[1];
    float4 a0 = sa[0], a1 = sa[1];
    int tb = y * 8;
    Xdem[tb+0][c]=d0.x; Xdem[tb+1][c]=d0.y; Xdem[tb+2][c]=d0.z; Xdem[tb+3][c]=d0.w;
    Xdem[tb+4][c]=d1.x; Xdem[tb+5][c]=d1.y; Xdem[tb+6][c]=d1.z; Xdem[tb+7][c]=d1.w;
    Xae [tb+0][c]=a0.x; Xae [tb+1][c]=a0.y; Xae [tb+2][c]=a0.z; Xae [tb+3][c]=a0.w;
    Xae [tb+4][c]=a1.x; Xae [tb+5][c]=a1.y; Xae [tb+6][c]=a1.z; Xae [tb+7][c]=a1.w;
  }
  if (t < 32) {
    int c = t >> 3, y = t & 7;
    const float4* st = (const float4*)(trust + (((b * 4 + c) * H + hy + y) * W + hx));
    float4 t0 = st[0], t1 = st[1];
    int tb = y * 8;
    Xtr[tb+0][c]=t0.x; Xtr[tb+1][c]=t0.y; Xtr[tb+2][c]=t0.z; Xtr[tb+3][c]=t0.w;
    Xtr[tb+4][c]=t1.x; Xtr[tb+5][c]=t1.y; Xtr[tb+6][c]=t1.z; Xtr[tb+7][c]=t1.w;
  }
  if (t < 64) { Xtr[t][4]=0.f; Xtr[t][5]=0.f; Xtr[t][6]=0.f; Xtr[t][7]=0.f; }
  __syncthreads();

  // ---------------- phase 1: q/k/v projections (M=tok, N=out-ch, K=64) ----------
  {
    const float* Wmats[3] = {Wq, Wk, Wv};
#pragma unroll
    for (int p = 0; p < 3; ++p) {
      const float* S = (p == 0) ? &Xdem[0][0] : &Xae[0][0];
      _Float16*    D = (p == 0) ? &Qs[0][0] : ((p == 1) ? &Ksm[0][0] : &Vsm[0][0]);
#pragma unroll
      for (int i = 0; i < 2; ++i) {
        int tile = wave + 8 * i;         // 16 MN tiles over 8 waves
        int mt = tile >> 2, nt = tile & 3;
        v8f acc = {};
#pragma unroll
        for (int ks = 0; ks < 64; ks += 32) {
          v16h a  = frag_a_f(S, LDX, mt * 16, ks, lane);
          v16h bb = frag_b_w(Wmats[p], 64, nt * 16, ks, lane);
          acc = wmma_f16(a, bb, acc);
        }
        int n  = nt * 16 + (lane & 15);
        int mB = mt * 16 + ((lane >= 16) ? 8 : 0);
#pragma unroll
        for (int r = 0; r < 8; ++r) D[(mB + r) * LDH + n] = (_Float16)acc[r];
      }
    }
  }
  __syncthreads();

  // ---------------- phase 2+3: attention (2 waves per head) ---------------------
  {
    const int h = wave >> 1;
#pragma unroll
    for (int mi = 0; mi < 2; ++mi) {
      int mt = (wave & 1) * 2 + mi;      // this wave's query M-tile
      v8f fr[4];
      // logits: K=head_dim 16 zero-padded to 32
#pragma unroll
      for (int nt = 0; nt < 4; ++nt) {
        v16h a;
        { int m = mt * 16 + (lane & 15); int k0 = (lane >= 16) ? 16 : 0;
#pragma unroll
          for (int j = 0; j < 16; ++j) {
            int kk = k0 + j;
            a[j] = (kk < 16) ? Qs[m][h * 16 + kk] : (_Float16)0.f;
          } }
        v16h bb;
        { int n = nt * 16 + (lane & 15); int k0 = (lane >= 16) ? 16 : 0;
#pragma unroll
          for (int j = 0; j < 16; ++j) {
            int kk = k0 + j;
            bb[j] = (kk < 16) ? Ksm[n][h * 16 + kk] : (_Float16)0.f;
          } }
        v8f z = {};
        fr[nt] = wmma_f16(a, bb, z);
      }
      // in-register softmax over each row of 64 (4 frags x 16 lanes per half-group)
#pragma unroll
      for (int r = 0; r < 8; ++r) {
        float x0 = fr[0][r] * 0.25f, x1 = fr[1][r] * 0.25f;
        float x2 = fr[2][r] * 0.25f, x3 = fr[3][r] * 0.25f;
        float mx = fmaxf(fmaxf(x0, x1), fmaxf(x2, x3));
        mx = fmaxf(mx, __shfl_xor(mx, 1, 32));
        mx = fmaxf(mx, __shfl_xor(mx, 2, 32));
        mx = fmaxf(mx, __shfl_xor(mx, 4, 32));
        mx = fmaxf(mx, __shfl_xor(mx, 8, 32));
        float e0 = __expf(x0 - mx), e1 = __expf(x1 - mx);
        float e2 = __expf(x2 - mx), e3 = __expf(x3 - mx);
        float s = e0 + e1 + e2 + e3;
        s += __shfl_xor(s, 1, 32);
        s += __shfl_xor(s, 2, 32);
        s += __shfl_xor(s, 4, 32);
        s += __shfl_xor(s, 8, 32);
        float inv = 1.0f / s;
        int row = mt * 16 + r + ((lane >= 16) ? 8 : 0);
        int col = lane & 15;
        Attn[h][row][ 0 + col] = (_Float16)(e0 * inv);
        Attn[h][row][16 + col] = (_Float16)(e1 * inv);
        Attn[h][row][32 + col] = (_Float16)(e2 * inv);
        Attn[h][row][48 + col] = (_Float16)(e3 * inv);
      }
      // AV: M=tok, N=head_dim 16, K=64 keys (intra-wave LDS RAW, dscnt-handled)
      v8f oacc = {};
#pragma unroll
      for (int ks = 0; ks < 64; ks += 32) {
        v16h a = frag_a_h(&Attn[h][0][0], LDH, mt * 16, ks, lane);
        v16h bb;
        { int n = lane & 15; int k0 = ks + ((lane >= 16) ? 16 : 0);
#pragma unroll
          for (int j = 0; j < 16; ++j) bb[j] = Vsm[k0 + j][h * 16 + n]; }
        oacc = wmma_f16(a, bb, oacc);
      }
      { int n = h * 16 + (lane & 15); int mB = mt * 16 + ((lane >= 16) ? 8 : 0);
#pragma unroll
        for (int r = 0; r < 8; ++r) Osm[mB + r][n] = (_Float16)oacc[r]; }
    }
  }
  __syncthreads();

  // ---------------- phase 4: a = o @ Wo^T  (M=tok, N=oc, K=64) -----------------
#pragma unroll
  for (int i = 0; i < 2; ++i) {
    int tile = wave + 8 * i;
    int mt = tile >> 2, nt = tile & 3;
    v8f acc = {};
#pragma unroll
    for (int ks = 0; ks < 64; ks += 32) {
      v16h a  = frag_a_h(&Osm[0][0], LDH, mt * 16, ks, lane);
      v16h bb = frag_b_w(Wo, 64, nt * 16, ks, lane);
      acc = wmma_f16(a, bb, acc);
    }
    int n  = nt * 16 + (lane & 15);
    int mB = mt * 16 + ((lane >= 16) ? 8 : 0);
#pragma unroll
    for (int r = 0; r < 8; ++r) Aa[mB + r][n] = acc[r];
  }
  __syncthreads();

  // ---------------- phase 5: gate GEMM (K = 64 dem + 64 a + 4 trust) + output --
  {
    const float alphaF = (float)alphaP[0];
#pragma unroll
    for (int i = 0; i < 2; ++i) {
      int tile = wave + 8 * i;
      int mt = tile >> 2, nt = tile & 3;
      v8f acc = {};
#pragma unroll
      for (int ks = 0; ks < 64; ks += 32) {              // chunks 0,1: f_dem
        v16h a  = frag_a_f(&Xdem[0][0], LDX, mt * 16, ks, lane);
        v16h bb = frag_b_w(Wg, 132, nt * 16, ks, lane);
        acc = wmma_f16(a, bb, acc);
      }
#pragma unroll
      for (int ks = 0; ks < 64; ks += 32) {              // chunks 2,3: a
        v16h a  = frag_a_f(&Aa[0][0], LDX, mt * 16, ks, lane);
        v16h bb = frag_b_w(Wg, 132, nt * 16, 64 + ks, lane);
        acc = wmma_f16(a, bb, acc);
      }
      {                                                  // chunk 4: trust (4 -> pad 32)
        v16h a;
        { int m = mt * 16 + (lane & 15); int k0 = (lane >= 16) ? 16 : 0;
#pragma unroll
          for (int j = 0; j < 16; ++j) {
            int kk = k0 + j;
            a[j] = (kk < 8) ? (_Float16)Xtr[m][kk] : (_Float16)0.f;  // cols 4..7 are zeroed
          } }
        v16h bb;
        { int n = nt * 16 + (lane & 15); int k0 = (lane >= 16) ? 16 : 0;
#pragma unroll
          for (int j = 0; j < 16; ++j) {
            int kk = k0 + j;
            bb[j] = (kk < 4) ? (_Float16)Wg[n * 132 + 128 + kk] : (_Float16)0.f;
          } }
        acc = wmma_f16(a, bb, acc);
      }
      // sigmoid gate + residual, stage into Xae as [oc][tok] for coalesced store
      int n  = nt * 16 + (lane & 15);
      int mB = mt * 16 + ((lane >= 16) ? 8 : 0);
      float bgv = bg[n];
#pragma unroll
      for (int r = 0; r < 8; ++r) {
        int tok = mB + r;
        float g  = 1.0f / (1.0f + __expf(-(acc[r] + bgv)));
        float av = Aa[tok][n];
        float fd = Xdem[tok][n];
        Xae[n][tok] = fd + alphaF * g * av;
      }
    }
  }
  __syncthreads();

  // ---------------- write out (coalesced 32B rows per channel) ------------------
#pragma unroll
  for (int i = 0; i < 2; ++i) {
    int job = t + 256 * i;
    int c = job >> 3, y = job & 7;
    float4* dst = (float4*)(out + (((b * 64 + c) * H + hy + y) * W + hx));
    int tb = y * 8;
    float4 v0 = { Xae[c][tb+0], Xae[c][tb+1], Xae[c][tb+2], Xae[c][tb+3] };
    float4 v1 = { Xae[c][tb+4], Xae[c][tb+5], Xae[c][tb+6], Xae[c][tb+7] };
    dst[0] = v0;
    dst[1] = v1;
  }
}

extern "C" void kernel_launch(void* const* d_in, const int* in_sizes, int n_in,
                              void* d_out, int out_size, void* d_ws, size_t ws_size,
                              hipStream_t stream) {
  (void)in_sizes; (void)n_in; (void)out_size; (void)d_ws; (void)ws_size;
  const float* f_dem = (const float*)d_in[0];
  const float* f_ae  = (const float*)d_in[1];
  const float* trust = (const float*)d_in[2];
  const float* Wq    = (const float*)d_in[3];
  const float* Wk    = (const float*)d_in[4];
  const float* Wv    = (const float*)d_in[5];
  const float* Wo    = (const float*)d_in[6];
  const float* Wg    = (const float*)d_in[7];
  const float* bg    = (const float*)d_in[8];
  const int*   alpha = (const int*)d_in[9];
  float* out = (float*)d_out;

  // 8 batches x 32 x 32 windows = 8192 workgroups, 8 wave32 each
  wca_fused_gfx1250<<<dim3(8192), dim3(256), 0, stream>>>(
      f_dem, f_ae, trust, Wq, Wk, Wv, Wo, Wg, bg, alpha, out);
}
